// LinkPredictor_69518340653680
// MI455X (gfx1250) — matrix operand order repaired and checked by reading, not compiled, not run on hardware
//
#include <hip/hip_runtime.h>

// ---------------------------------------------------------------------------
// Types for CDNA5 WMMA (wave32)
// ---------------------------------------------------------------------------
typedef __attribute__((ext_vector_type(16))) __bf16         v16bf;
typedef __attribute__((ext_vector_type(16))) unsigned short v16u;
typedef __attribute__((ext_vector_type(8)))  float          v8f;

__device__ __forceinline__ unsigned short f2bf(float f) {
  union { float f; unsigned u; } c; c.f = f;
  unsigned u = c.u;
  u += 0x7FFFu + ((u >> 16) & 1u);   // round-to-nearest-even to bf16
  return (unsigned short)(u >> 16);
}
__device__ __forceinline__ float bf2f(unsigned short h) {
  union { unsigned u; float f; } c; c.u = ((unsigned)h) << 16;
  return c.f;
}
__device__ __forceinline__ void split_bf(float f, unsigned short& hi, unsigned short& lo) {
  hi = f2bf(f);
  lo = f2bf(f - bf2f(hi));
}
__device__ __forceinline__ void split4(float4 v, ushort4& h, ushort4& l) {
  split_bf(v.x, h.x, l.x);
  split_bf(v.y, h.y, l.y);
  split_bf(v.z, h.z, l.z);
  split_bf(v.w, h.w, l.w);
}

#define WMMA_BF16(a, b, c) \
  __builtin_amdgcn_wmma_f32_16x16x32_bf16(false, (a), false, (b), (short)0, (c), false, false)

// Fragment-order inverse maps (wave32 16x16x32 bf16 WMMA):
//   A element (row r, kk=k&31): lane = ((kk>>3)&1)*16 + r, i = ((kk>>4)<<3)|(kk&7)
//   B element (kk=k&31, col n): lane = ((kk>>3)&1)*16 + (n&15), i = ((kk>>4)<<3)|(kk&7)

// ---------------------------------------------------------------------------
// Degree / normalization kernels
// ---------------------------------------------------------------------------
__global__ void deg_init(float* __restrict__ deg, int N) {
  int i = blockIdx.x * blockDim.x + threadIdx.x;
  if (i < N) deg[i] = 1.0f;                      // self-loop
}
__global__ void deg_accum(float* __restrict__ deg, const int* __restrict__ dst, int E) {
  int i = blockIdx.x * blockDim.x + threadIdx.x;
  if (i < E) atomicAdd(&deg[dst[i]], 1.0f);
}
__global__ void deg_rsqrt(float* __restrict__ deg, int N) {
  int i = blockIdx.x * blockDim.x + threadIdx.x;
  if (i < N) {
    float d = deg[i];
    deg[i] = (d > 0.0f) ? rsqrtf(d) : 0.0f;
  }
}

// ---------------------------------------------------------------------------
// Zero a buffer (float4 granularity)
// ---------------------------------------------------------------------------
__global__ void zero_f4(float4* __restrict__ p, long n4) {
  long i = (long)blockIdx.x * blockDim.x + threadIdx.x;
  if (i < n4) p[i] = make_float4(0.f, 0.f, 0.f, 0.f);
}

// ---------------------------------------------------------------------------
// GEMM: Y[N,128] = A[N,128] @ W[128,128]  via 3xBF16 compensated WMMA.
// Block = 256 threads = 8 waves; wave w owns column tile w.
// LDS staged in WMMA *fragment order*: each lane's 16 bf16 = contiguous 32B.
// ---------------------------------------------------------------------------
__global__ __launch_bounds__(256) void gemm_n128(const float* __restrict__ A,
                                                 const float* __restrict__ W,
                                                 float* __restrict__ Y, int N) {
  __shared__ __align__(32) unsigned short sAh[4][32][16];  // [ks][lane][i]
  __shared__ __align__(32) unsigned short sAl[4][32][16];
  __shared__ __align__(32) unsigned short sBh[8][32][16];  // [tile][lane][i] (one ks slice)
  __shared__ __align__(32) unsigned short sBl[8][32][16];

  const int tid = threadIdx.x;
  const long rowBase = (long)blockIdx.x * 16;

  // Stage full A tile (16 rows x 128 K) in fragment order, float4 granularity.
  for (int u = tid; u < 16 * 32; u += 256) {
    int r = u >> 5, k = (u & 31) * 4;
    long row = rowBase + r;
    if (row >= N) row = N - 1;
    float4 a = *(const float4*)(A + row * 128 + k);
    int ks = k >> 5, kk = k & 31;
    int hi = (kk >> 3) & 1;
    int ib = ((kk >> 4) << 3) | (kk & 7);
    int lane = hi * 16 + r;
    ushort4 h, l;
    split4(a, h, l);
    *(ushort4*)&sAh[ks][lane][ib] = h;    // 4 consecutive i -> contiguous store
    *(ushort4*)&sAl[ks][lane][ib] = l;
  }

  const int wave = tid >> 5;          // column tile 0..7
  const int lane = tid & 31;
  const int l16  = lane & 15;
  const int hi   = lane >> 4;

  v8f acc = {};
#pragma unroll
  for (int ks = 0; ks < 4; ++ks) {
    __syncthreads();
    // Stage W K-slice rows [ks*32, ks*32+32) in fragment order.
    for (int u = tid; u < 32 * 32; u += 256) {
      int kk = u >> 5, n = (u & 31) * 4;
      float4 w = *(const float4*)(W + (long)(ks * 32 + kk) * 128 + n);
      int bhi = (kk >> 3) & 1, bi = ((kk >> 4) << 3) | (kk & 7);
      int tile = n >> 4, bl = bhi * 16 + (n & 15);
      ushort4 h, l;
      split4(w, h, l);
      sBh[tile][bl + 0][bi] = h.x; sBl[tile][bl + 0][bi] = l.x;
      sBh[tile][bl + 1][bi] = h.y; sBl[tile][bl + 1][bi] = l.y;
      sBh[tile][bl + 2][bi] = h.z; sBl[tile][bl + 2][bi] = l.z;
      sBh[tile][bl + 3][bi] = h.w; sBl[tile][bl + 3][bi] = l.w;
    }
    __syncthreads();

    v16bf ahb = __builtin_bit_cast(v16bf, *(const v16u*)&sAh[ks][lane][0]);
    v16bf alb = __builtin_bit_cast(v16bf, *(const v16u*)&sAl[ks][lane][0]);
    v16bf bhb = __builtin_bit_cast(v16bf, *(const v16u*)&sBh[wave][lane][0]);
    v16bf blb = __builtin_bit_cast(v16bf, *(const v16u*)&sBl[wave][lane][0]);

    acc = WMMA_BF16(ahb, bhb, acc);
    acc = WMMA_BF16(ahb, blb, acc);
    acc = WMMA_BF16(alb, bhb, acc);
  }

#pragma unroll
  for (int r = 0; r < 8; ++r) {
    long row = rowBase + hi * 8 + r;
    if (row < N) Y[row * 128 + wave * 16 + l16] = acc[r];
  }
}

// ---------------------------------------------------------------------------
// Edge scatter: agg[dst] += H[src] * dinv[src]*dinv[dst]
// 32 lanes per edge, float4 per lane.
// ---------------------------------------------------------------------------
__global__ void scatter_edges(const float* __restrict__ H,
                              const int* __restrict__ src,
                              const int* __restrict__ dst,
                              const float* __restrict__ dinv,
                              float* __restrict__ agg, int E) {
  long t = (long)blockIdx.x * blockDim.x + threadIdx.x;
  long e = t >> 5;
  int lane = (int)(t & 31);
  if (e >= E) return;
  int s = src[e], d = dst[e];
  float norm = dinv[s] * dinv[d];
  float4 v = ((const float4*)(H + (long)s * 128))[lane];
  float* arow = agg + (long)d * 128 + lane * 4;
  atomicAdd(arow + 0, v.x * norm);
  atomicAdd(arow + 1, v.y * norm);
  atomicAdd(arow + 2, v.z * norm);
  atomicAdd(arow + 3, v.w * norm);
}

// ---------------------------------------------------------------------------
// Epilogue: out = [relu](agg + H[n]*dinv[n]^2 + bias)  (in place on agg)
// float4 granularity.
// ---------------------------------------------------------------------------
__global__ void node_epilogue(float* __restrict__ agg,
                              const float* __restrict__ H,
                              const float* __restrict__ dinv,
                              const float* __restrict__ bias,
                              int N, int doRelu) {
  long t = (long)blockIdx.x * blockDim.x + threadIdx.x;   // one float4 each
  long n = t >> 5;
  int f = (int)(t & 31) * 4;
  if (n >= N) return;
  float di = dinv[n];
  float d2 = di * di;
  float4 a = *(float4*)(agg + n * 128 + f);
  float4 h = *(const float4*)(H + n * 128 + f);
  float4 b = *(const float4*)(bias + f);
  float4 v;
  v.x = a.x + h.x * d2 + b.x;
  v.y = a.y + h.y * d2 + b.y;
  v.z = a.z + h.z * d2 + b.z;
  v.w = a.w + h.w * d2 + b.w;
  if (doRelu) {
    v.x = fmaxf(v.x, 0.f); v.y = fmaxf(v.y, 0.f);
    v.z = fmaxf(v.z, 0.f); v.w = fmaxf(v.w, 0.f);
  }
  *(float4*)(agg + n * 128 + f) = v;
}

// ---------------------------------------------------------------------------
// Decoder: rows = pos edges then neg edges.
//   feat[r] = [z[u], z[v]] (256); hd = relu(feat @ Wd1 + bd1); out = hd.Wd2 + bd2
// Block = 256 threads = 8 waves handles 16 rows; wave w owns col tiles {w, w+8}.
// LDS staged in fragment order (contiguous 32B per lane fragment).
// ---------------------------------------------------------------------------
__global__ __launch_bounds__(256) void decoder_kernel(const float* __restrict__ Z,
                                                      const int* __restrict__ pos,
                                                      const int* __restrict__ neg,
                                                      const float* __restrict__ Wd1,
                                                      const float* __restrict__ bd1,
                                                      const float* __restrict__ Wd2,
                                                      const float* __restrict__ bd2,
                                                      float* __restrict__ out,
                                                      int Ep, int rows) {
  __shared__ __align__(32) unsigned short sFh[8][32][16];   // [ks][lane][i]
  __shared__ __align__(32) unsigned short sFl[8][32][16];
  __shared__ __align__(32) unsigned short sWh[16][32][16];  // [tile][lane][i] (one ks slice)
  __shared__ __align__(32) unsigned short sWl[16][32][16];
  __shared__ float sPart[16];

  const int tid = threadIdx.x;
  const long rowBase = (long)blockIdx.x * 16;

  // Stage gathered feature tile (16 x 256) in fragment order.
  for (int u = tid; u < 16 * 64; u += 256) {
    int r = u >> 6, c = (u & 63) * 4;
    long R = rowBase + r;
    float4 v = make_float4(0.f, 0.f, 0.f, 0.f);
    if (R < rows) {
      const int* edges = (R < Ep) ? pos : neg;
      long idx = (R < Ep) ? R : (R - Ep);
      int node = (c < 128) ? edges[idx] : edges[(long)Ep + idx];
      v = *(const float4*)(Z + (long)node * 128 + (c & 127));
    }
    int ks = c >> 5, kk = c & 31;
    int fhi = (kk >> 3) & 1;
    int ib = ((kk >> 4) << 3) | (kk & 7);
    int lane = fhi * 16 + r;
    ushort4 h, l;
    split4(v, h, l);
    *(ushort4*)&sFh[ks][lane][ib] = h;
    *(ushort4*)&sFl[ks][lane][ib] = l;
  }
  if (tid < 16) sPart[tid] = 0.0f;

  const int wave = tid >> 5;
  const int lane = tid & 31;
  const int l16  = lane & 15;
  const int hi   = lane >> 4;
  const int t0 = wave, t1 = wave + 8;   // two 16-wide column tiles per wave

  v8f acc0 = {};
  v8f acc1 = {};
  for (int ks = 0; ks < 8; ++ks) {
    __syncthreads();
    // Stage Wd1 K-slice rows [ks*32, ks*32+32) x 256 cols in fragment order.
    for (int u = tid; u < 32 * 64; u += 256) {
      int kk = u >> 6, n = (u & 63) * 4;
      float4 w = *(const float4*)(Wd1 + (long)(ks * 32 + kk) * 256 + n);
      int bhi = (kk >> 3) & 1, bi = ((kk >> 4) << 3) | (kk & 7);
      int tile = n >> 4, bl = bhi * 16 + (n & 15);
      ushort4 h, l;
      split4(w, h, l);
      sWh[tile][bl + 0][bi] = h.x; sWl[tile][bl + 0][bi] = l.x;
      sWh[tile][bl + 1][bi] = h.y; sWl[tile][bl + 1][bi] = l.y;
      sWh[tile][bl + 2][bi] = h.z; sWl[tile][bl + 2][bi] = l.z;
      sWh[tile][bl + 3][bi] = h.w; sWl[tile][bl + 3][bi] = l.w;
    }
    // Prefetch next K-slice of Wd1 while this slice computes.
    if (ks < 7) __builtin_prefetch(Wd1 + (long)(ks + 1) * 32 * 256 + tid * 32, 0, 3);
    __syncthreads();

    v16bf ahb = __builtin_bit_cast(v16bf, *(const v16u*)&sFh[ks][lane][0]);
    v16bf alb = __builtin_bit_cast(v16bf, *(const v16u*)&sFl[ks][lane][0]);
    v16bf bhb0 = __builtin_bit_cast(v16bf, *(const v16u*)&sWh[t0][lane][0]);
    v16bf blb0 = __builtin_bit_cast(v16bf, *(const v16u*)&sWl[t0][lane][0]);
    v16bf bhb1 = __builtin_bit_cast(v16bf, *(const v16u*)&sWh[t1][lane][0]);
    v16bf blb1 = __builtin_bit_cast(v16bf, *(const v16u*)&sWl[t1][lane][0]);

    acc0 = WMMA_BF16(ahb, bhb0, acc0);
    acc0 = WMMA_BF16(ahb, blb0, acc0);
    acc0 = WMMA_BF16(alb, bhb0, acc0);
    acc1 = WMMA_BF16(ahb, bhb1, acc1);
    acc1 = WMMA_BF16(ahb, blb1, acc1);
    acc1 = WMMA_BF16(alb, bhb1, acc1);
  }

  // Fused bias + ReLU + projection onto Wd2, reduced into per-row LDS partials
  {
    int col0 = t0 * 16 + l16;
    int col1 = t1 * 16 + l16;
    float w0 = Wd2[col0], bb0 = bd1[col0];
    float w1 = Wd2[col1], bb1 = bd1[col1];
#pragma unroll
    for (int r = 0; r < 8; ++r) {
      int row = hi * 8 + r;
      float v0 = fmaxf(acc0[r] + bb0, 0.0f);
      float v1 = fmaxf(acc1[r] + bb1, 0.0f);
      atomicAdd(&sPart[row], v0 * w0 + v1 * w1);
    }
  }
  __syncthreads();
  if (tid < 16) {
    long R = rowBase + tid;
    if (R < rows) out[R] = sPart[tid] + bd2[0];
  }
}

// ---------------------------------------------------------------------------
// Launch
// ---------------------------------------------------------------------------
extern "C" void kernel_launch(void* const* d_in, const int* in_sizes, int n_in,
                              void* d_out, int out_size, void* d_ws, size_t ws_size,
                              hipStream_t stream) {
  const float* x   = (const float*)d_in[0];
  const int*   ei  = (const int*)  d_in[1];
  const int*   pos = (const int*)  d_in[2];
  const int*   neg = (const int*)  d_in[3];
  const float* W1  = (const float*)d_in[4];
  const float* b1  = (const float*)d_in[5];
  const float* W2  = (const float*)d_in[6];
  const float* b2  = (const float*)d_in[7];
  const float* Wd1 = (const float*)d_in[8];
  const float* bd1 = (const float*)d_in[9];
  const float* Wd2 = (const float*)d_in[10];
  const float* bd2 = (const float*)d_in[11];
  float* out = (float*)d_out;

  const int N  = in_sizes[0] / 128;
  const int E  = in_sizes[1] / 2;
  const int Ep = in_sizes[2] / 2;

  char* ws = (char*)d_ws;
  size_t off = 0;
  float* dinv = (float*)(ws + off); off += ((size_t)N * 4 + 255) & ~(size_t)255;
  float* buf0 = (float*)(ws + off); off += (size_t)N * 128 * 4;
  float* buf1 = (float*)(ws + off);

  const long n4 = (long)N * 128 / 4;     // float4 count
  const long nt = (long)N * 32;          // epilogue threads (float4 each)
  const long st = (long)E * 32;          // scatter thread count
  const int rowTiles = (N + 15) / 16;
  const int rows = 2 * Ep;

  // symmetric normalization dinv = rsqrt(deg+1)
  deg_init <<<(N + 255) / 256, 256, 0, stream>>>(dinv, N);
  deg_accum<<<(E + 255) / 256, 256, 0, stream>>>(dinv, ei + E, E);
  deg_rsqrt<<<(N + 255) / 256, 256, 0, stream>>>(dinv, N);

  // Layer 1: h = relu(aggregate(x @ W1) + b1)
  gemm_n128<<<rowTiles, 256, 0, stream>>>(x, W1, buf0, N);
  zero_f4<<<(int)((n4 + 255) / 256), 256, 0, stream>>>((float4*)buf1, n4);
  scatter_edges<<<(int)((st + 255) / 256), 256, 0, stream>>>(buf0, ei, ei + E, dinv, buf1, E);
  node_epilogue<<<(int)((nt + 255) / 256), 256, 0, stream>>>(buf1, buf0, dinv, b1, N, 1);

  // Layer 2: z = aggregate(h @ W2) + b2
  gemm_n128<<<rowTiles, 256, 0, stream>>>(buf1, W2, buf0, N);
  zero_f4<<<(int)((n4 + 255) / 256), 256, 0, stream>>>((float4*)buf1, n4);
  scatter_edges<<<(int)((st + 255) / 256), 256, 0, stream>>>(buf0, ei, ei + E, dinv, buf1, E);
  node_epilogue<<<(int)((nt + 255) / 256), 256, 0, stream>>>(buf1, buf0, dinv, b2, N, 0);

  // Decoder over pos||neg edge rows
  decoder_kernel<<<(rows + 15) / 16, 256, 0, stream>>>(buf1, pos, neg, Wd1, bd1, Wd2, bd2,
                                                       out, Ep, rows);
}
